// ChannelDiffusion_970662609223
// MI455X (gfx1250) — compile-verified, matrix operand order checked by reading, not run on hardware
//
#include <hip/hip_runtime.h>
#include <hip/hip_bf16.h>
#include <stdint.h>

// ---------------------------------------------------------------------------
// CDNA5 (gfx1250) wave32 WMMA implementation of ChannelDiffusion.
//   B=8, N=4096, D=1024, H=16, DH=64.
//   Heavy math: v_wmma_f32_16x16x32_bf16, f32 accumulation.
//   Big GEMMs: double-buffered LDS staged with GLOBAL_LOAD_ASYNC_TO_LDS_B128
//   (ASYNCcnt-tracked DMA, no VGPR round trip), s_wait_asynccnt pipelining.
// ---------------------------------------------------------------------------

typedef __attribute__((ext_vector_type(16))) __bf16 v16bf;
typedef __attribute__((ext_vector_type(8)))  float  v8f;

#define WMMA_BF16(a, b, c) \
  __builtin_amdgcn_wmma_f32_16x16x32_bf16(false, (a), false, (b), (short)0, (c), false, false)

// Async DMA: global -> LDS, 16 bytes per lane.  VDST VGPR carries the LDS
// byte offset (low 32 bits of the generic shared pointer), VADDR the 64-bit
// global address.  Tracked by ASYNCcnt; completes in order.
#define ASYNC_LD_B128(ldsOff, gptr)                                         \
  asm volatile("global_load_async_to_lds_b128 %0, %1, off"                  \
               :: "v"(ldsOff), "v"(gptr) : "memory")
#define ASYNC_WAIT(n) asm volatile("s_wait_asynccnt " #n ::: "memory")

__device__ __forceinline__ uint32_t lds_off_u32(const void* p) {
  return (uint32_t)(uintptr_t)p;   // flat shared aperture: low 32 bits = DS offset
}

__device__ __forceinline__ uint16_t f32_to_bf16(float f) {
  union { float f; uint32_t u; } v; v.f = f;
  uint32_t u = v.u;
  u += 0x7FFFu + ((u >> 16) & 1u);   // round to nearest even
  return (uint16_t)(u >> 16);
}

__device__ __forceinline__ uint32_t pack2_bf16(float lo, float hi) {
  return (uint32_t)f32_to_bf16(lo) | ((uint32_t)f32_to_bf16(hi) << 16);
}

// Load a 16x32 bf16 A-fragment from a row-major tile (row stride in elems).
// ISA 7.12.2 layout: lane m = l&15; k-base = (l<16)?0:8; dwords 0..3 hold
// k = kb+2j, dwords 4..7 hold k = 16+kb+2j (pairs contiguous in memory).
// B fragments use the same function on the transposed B tile (Bt = B^T).
__device__ __forceinline__ v16bf load_frag(const uint16_t* base, int stride, int lane) {
  const int m  = lane & 15;
  const int kb = (lane < 16) ? 0 : 8;
  union { v16bf v; uint32_t u[8]; } f;
  const uint16_t* row = base + (size_t)m * stride;
#pragma unroll
  for (int j = 0; j < 4; ++j) {
    f.u[j]     = *(const uint32_t*)(row + kb + 2 * j);
    f.u[j + 4] = *(const uint32_t*)(row + 16 + kb + 2 * j);
  }
  return f.v;
}

// ---------------------------------------------------------------------------
// 1) f32 -> bf16 bulk convert: 8 elements / thread, b128 in, b128 out.
// ---------------------------------------------------------------------------
__global__ __launch_bounds__(256)
void cvt_bf16_kernel(const float* __restrict__ in, uint16_t* __restrict__ out,
                     int n8) {
  const int i = blockIdx.x * 256 + threadIdx.x;
  if (i >= n8) return;
  const float4* in4 = (const float4*)in + (size_t)i * 2;
  float4 a = in4[0], b = in4[1];
  uint4 p;
  p.x = pack2_bf16(a.x, a.y);
  p.y = pack2_bf16(a.z, a.w);
  p.z = pack2_bf16(b.x, b.y);
  p.w = pack2_bf16(b.z, b.w);
  ((uint4*)out)[i] = p;
}

// ---------------------------------------------------------------------------
// 2) Weight transpose + convert: Bt[n][k] = bf16(W[k][n]), W is dim x dim.
// ---------------------------------------------------------------------------
__global__ __launch_bounds__(256)
void transpose_cvt_kernel(const float* __restrict__ W, uint16_t* __restrict__ Bt,
                          int dim) {
  __shared__ __attribute__((aligned(16))) uint16_t sT[64 * 65];
  const int tid = threadIdx.x;
  const int k0  = blockIdx.y * 64;
  const int n0  = blockIdx.x * 64;

  {  // load: row-major over n, convert, store [k][n]
    const int e  = tid * 16;
    const int kk = e >> 6;
    const int nn = e & 63;
    const float* src = W + (size_t)(k0 + kk) * dim + n0 + nn;
#pragma unroll
    for (int q = 0; q < 16; ++q) sT[kk * 65 + nn + q] = f32_to_bf16(src[q]);
  }
  __syncthreads();
  {  // store: row-major over k for fixed n, packed dwords
    const int e   = tid * 16;
    const int nn  = e >> 6;
    const int kk0 = e & 63;
    uint32_t p[8];
#pragma unroll
    for (int q = 0; q < 8; ++q) {
      uint32_t lo = sT[(kk0 + 2 * q) * 65 + nn];
      uint32_t hi = sT[(kk0 + 2 * q + 1) * 65 + nn];
      p[q] = lo | (hi << 16);
    }
    uint32_t* dst = (uint32_t*)(Bt + (size_t)(n0 + nn) * dim + kk0);
#pragma unroll
    for (int q = 0; q < 8; ++q) dst[q] = p[q];
  }
}

// ---------------------------------------------------------------------------
// 3) GEMM: C[M x Nout] = A[M x K] * B, Bt = B^T row-major (N x K), bf16 in.
//    256 threads / 8 waves, tile 128x128, K_TILE = 32.
//    Wave (wm,wn) computes 64x32 = 4x2 WMMA tiles.
//    Double-buffered LDS filled by async-to-LDS DMA; per tile each thread
//    issues 4 b128 async loads; steady-state s_wait_asynccnt 4 drains the
//    current buffer while the next tile's DMA stays in flight.
// ---------------------------------------------------------------------------
template <bool STORE_F32>
__global__ __launch_bounds__(256)
void gemm_wmma_kernel(const uint16_t* __restrict__ A,
                      const uint16_t* __restrict__ Bt,
                      void* __restrict__ Cv, int M, int Nout, int K) {
  __shared__ __attribute__((aligned(16))) uint16_t sA[2][128 * 32];
  __shared__ __attribute__((aligned(16))) uint16_t sB[2][128 * 32];

  const int tid  = threadIdx.x;
  const int lane = tid & 31;
  const int wave = tid >> 5;
  const int wm   = wave >> 2;      // 0..1 : 64-row half
  const int wn   = wave & 3;       // 0..3 : 32-col slice
  const int bm   = blockIdx.y * 128;
  const int bn   = blockIdx.x * 128;

  // staging map: this thread fills rows r0 and r1, cols c0..c0+7 (one b128 each)
  const int r0 = tid >> 2;
  const int c0 = (tid & 3) * 8;
  const int r1 = 64 + r0;

  const uint16_t* Abase = A  + (size_t)bm * K;
  const uint16_t* Bbase = Bt + (size_t)bn * K;

  const uint32_t offA0[2] = { lds_off_u32(&sA[0][r0 * 32 + c0]),
                              lds_off_u32(&sA[1][r0 * 32 + c0]) };
  const uint32_t offA1[2] = { lds_off_u32(&sA[0][r1 * 32 + c0]),
                              lds_off_u32(&sA[1][r1 * 32 + c0]) };
  const uint32_t offB0[2] = { lds_off_u32(&sB[0][r0 * 32 + c0]),
                              lds_off_u32(&sB[1][r0 * 32 + c0]) };
  const uint32_t offB1[2] = { lds_off_u32(&sB[0][r1 * 32 + c0]),
                              lds_off_u32(&sB[1][r1 * 32 + c0]) };

  v8f acc[4][2];
  const v8f zacc = {0.f, 0.f, 0.f, 0.f, 0.f, 0.f, 0.f, 0.f};
#pragma unroll
  for (int i = 0; i < 4; ++i)
#pragma unroll
    for (int j = 0; j < 2; ++j) acc[i][j] = zacc;

  const int nk = K >> 5;

  // issue tile 0 into buffer 0
  {
    ASYNC_LD_B128(offA0[0], Abase + (size_t)r0 * K + c0);
    ASYNC_LD_B128(offA1[0], Abase + (size_t)r1 * K + c0);
    ASYNC_LD_B128(offB0[0], Bbase + (size_t)r0 * K + c0);
    ASYNC_LD_B128(offB1[0], Bbase + (size_t)r1 * K + c0);
  }

  for (int kt = 0; kt < nk; ++kt) {
    const int cur = kt & 1;
    if (kt + 1 < nk) {   // DMA next tile into the other buffer
      const int nb = cur ^ 1;
      const int ko = (kt + 1) * 32;
      ASYNC_LD_B128(offA0[nb], Abase + (size_t)r0 * K + ko + c0);
      ASYNC_LD_B128(offA1[nb], Abase + (size_t)r1 * K + ko + c0);
      ASYNC_LD_B128(offB0[nb], Bbase + (size_t)r0 * K + ko + c0);
      ASYNC_LD_B128(offB1[nb], Bbase + (size_t)r1 * K + ko + c0);
      ASYNC_WAIT(0x4);   // current tile's 4 loads (older, in-order) complete
    } else {
      ASYNC_WAIT(0x0);   // drain
    }
    __syncthreads();     // cross-wave: everyone's current-buffer DMA landed

    v16bf af[4];
#pragma unroll
    for (int i = 0; i < 4; ++i)
      af[i] = load_frag(&sA[cur][(wm * 64 + i * 16) * 32], 32, lane);
    v16bf bfr[2];
#pragma unroll
    for (int j = 0; j < 2; ++j)
      bfr[j] = load_frag(&sB[cur][(wn * 32 + j * 16) * 32], 32, lane);
#pragma unroll
    for (int i = 0; i < 4; ++i)
#pragma unroll
      for (int j = 0; j < 2; ++j)
        acc[i][j] = WMMA_BF16(af[i], bfr[j], acc[i][j]);

    __syncthreads();     // all waves done reading before this buffer is refilled
  }

  // C/D layout: VGPR r -> row r + 8*(lane>=16), col = lane&15
  const int colL = lane & 15;
  const int rowH = (lane >> 4) * 8;
#pragma unroll
  for (int i = 0; i < 4; ++i) {
#pragma unroll
    for (int j = 0; j < 2; ++j) {
      const int row0 = bm + wm * 64 + i * 16 + rowH;
      const int col  = bn + wn * 32 + j * 16 + colL;
#pragma unroll
      for (int r = 0; r < 8; ++r) {
        const size_t idx = (size_t)(row0 + r) * Nout + col;
        if (STORE_F32) ((float*)Cv)[idx] = acc[i][j][r];
        else           ((uint16_t*)Cv)[idx] = f32_to_bf16(acc[i][j][r]);
      }
    }
  }
}

// ---------------------------------------------------------------------------
// 4) Gram + softmax per (b,h).  QK lives in fused QKV (row stride 2048).
//    S[d][e] = sum_n QK[b,n,h*64+d]*QK[b,n,h*64+e];  q2 = diag(S).
//    logits = (2S[d][e]-S[d][d]-S[e][e]) * tau[h]/64; attn = softmax_e.
// ---------------------------------------------------------------------------
__global__ __launch_bounds__(256)
void gram_softmax_kernel(const uint16_t* __restrict__ QKV,
                         const float* __restrict__ tau,
                         uint16_t* __restrict__ attn) {
  __shared__ __attribute__((aligned(16))) uint16_t sT[64 * 32]; // [d][n]
  __shared__ float sS[64 * 65];
  __shared__ float sDiag[64];

  const int bh   = blockIdx.x;
  const int b    = bh >> 4;
  const int h    = bh & 15;
  const int tid  = threadIdx.x;
  const int lane = tid & 31;
  const int wave = tid >> 5;
  const int dt   = wave >> 1;           // d-tile 0..3
  const int et0  = (wave & 1) * 2;      // first e-tile

  const size_t rowBase = (size_t)b * 4096;
  const int    hOff    = h * 64;

  // staging: thread -> nn = tid>>3 (0..31), d0 = (tid&7)*8: one uint4 each
  const int nn = tid >> 3;
  const int d0 = (tid & 7) * 8;

  v8f acc[2];
  const v8f zacc = {0.f, 0.f, 0.f, 0.f, 0.f, 0.f, 0.f, 0.f};
  acc[0] = zacc; acc[1] = zacc;

  uint4 r = *(const uint4*)(QKV + (rowBase + nn) * 2048 + hOff + d0);
  for (int it = 0; it < 128; ++it) {
    {  // scatter 8 bf16 (consecutive d) into transposed tile
      union { uint4 v; uint16_t h[8]; } u; u.v = r;
#pragma unroll
      for (int q = 0; q < 8; ++q) sT[(d0 + q) * 32 + nn] = u.h[q];
    }
    __syncthreads();

    uint4 rn;
    if (it + 1 < 128)
      rn = *(const uint4*)(QKV + (rowBase + (it + 1) * 32 + nn) * 2048 + hOff + d0);

    v16bf a  = load_frag(sT + dt * 16 * 32, 32, lane);
    v16bf b0 = load_frag(sT + (et0 + 0) * 16 * 32, 32, lane);
    v16bf b1 = load_frag(sT + (et0 + 1) * 16 * 32, 32, lane);
    acc[0] = WMMA_BF16(a, b0, acc[0]);
    acc[1] = WMMA_BF16(a, b1, acc[1]);
    __syncthreads();
    r = rn;
  }

  const int colL = lane & 15;
  const int rowH = (lane >> 4) * 8;
#pragma unroll
  for (int j = 0; j < 2; ++j)
#pragma unroll
    for (int rr = 0; rr < 8; ++rr)
      sS[(dt * 16 + rowH + rr) * 65 + (et0 + j) * 16 + colL] = acc[j][rr];
  __syncthreads();

  if (tid < 64) sDiag[tid] = sS[tid * 65 + tid];
  __syncthreads();

  if (tid < 64) {
    const int d = tid;
    const float scale = tau[h] * (1.0f / 64.0f);   // tau / sqrt(N)
    const float dd = sDiag[d];
    float mx = -3.0e38f;
#pragma unroll 4
    for (int e = 0; e < 64; ++e) {
      float l = (2.0f * sS[d * 65 + e] - dd - sDiag[e]) * scale;
      sS[d * 65 + e] = l;
      mx = fmaxf(mx, l);
    }
    float sum = 0.0f;
#pragma unroll 4
    for (int e = 0; e < 64; ++e) {
      float p = __expf(sS[d * 65 + e] - mx);
      sS[d * 65 + e] = p;
      sum += p;
    }
    const float inv = 1.0f / sum;
    uint16_t* out = attn + ((size_t)bh * 64 + d) * 64;
#pragma unroll 4
    for (int e = 0; e < 64; ++e) out[e] = f32_to_bf16(sS[d * 65 + e] * inv);
  }
}

// ---------------------------------------------------------------------------
// 5) Y[b,n,h*64+d] = sum_e V[b,n,h*64+e] * attn[bh][d][e]   (K = 64)
//    V lives at QKV column offset 1024 (row stride 2048).  Fragments load
//    straight from global (k-pairs contiguous in both V rows and attn rows).
// ---------------------------------------------------------------------------
__global__ __launch_bounds__(256)
void apply_attn_kernel(const uint16_t* __restrict__ QKV,
                       const uint16_t* __restrict__ attn,
                       uint16_t* __restrict__ Y) {
  const int bh   = blockIdx.y;
  const int b    = bh >> 4;
  const int h    = bh & 15;
  const int n0   = blockIdx.x * 128;
  const int tid  = threadIdx.x;
  const int lane = tid & 31;
  const int wave = tid >> 5;

  const size_t rowA = ((size_t)b * 4096 + n0 + wave * 16);
  const uint16_t* Abase = QKV + rowA * 2048 + 1024 + h * 64;   // V block
  const uint16_t* Bbase = attn + (size_t)bh * 64 * 64;

  v8f acc[4];
  const v8f zacc = {0.f, 0.f, 0.f, 0.f, 0.f, 0.f, 0.f, 0.f};
#pragma unroll
  for (int j = 0; j < 4; ++j) acc[j] = zacc;

#pragma unroll
  for (int kc = 0; kc < 2; ++kc) {
    v16bf a = load_frag(Abase + kc * 32, 2048, lane);
#pragma unroll
    for (int j = 0; j < 4; ++j) {
      v16bf bb = load_frag(Bbase + (j * 16) * 64 + kc * 32, 64, lane);
      acc[j] = WMMA_BF16(a, bb, acc[j]);
    }
  }

  const int colL = lane & 15;
  const int rowH = (lane >> 4) * 8;
#pragma unroll
  for (int j = 0; j < 4; ++j)
#pragma unroll
    for (int r = 0; r < 8; ++r) {
      const size_t idx = (rowA + rowH + r) * 1024 + h * 64 + j * 16 + colL;
      Y[idx] = f32_to_bf16(acc[j][r]);
    }
}

// ---------------------------------------------------------------------------
// Host-side launcher
// ---------------------------------------------------------------------------
extern "C" void kernel_launch(void* const* d_in, const int* in_sizes, int n_in,
                              void* d_out, int out_size, void* d_ws, size_t ws_size,
                              hipStream_t stream) {
  const int B = 8, N = 4096, D = 1024, H = 16;
  const int M = B * N;                         // 32768 token rows

  const float* x   = (const float*)d_in[0];
  const float* Wqk = (const float*)d_in[1];
  const float* Wv  = (const float*)d_in[2];
  const float* Wo  = (const float*)d_in[3];
  const float* tau = (const float*)d_in[4];

  // workspace layout (bf16)
  char* ws = (char*)d_ws;
  size_t off = 0;
  uint16_t* QKV  = (uint16_t*)(ws + off); off += (size_t)M * 2 * D * 2;   // 128 MiB
  uint16_t* Ybuf = (uint16_t*)(ws + off); off += (size_t)M * D * 2;       //  64 MiB
  uint16_t* xb   = (uint16_t*)(ws + off); off += (size_t)M * D * 2;       //  64 MiB
  uint16_t* BtQV = (uint16_t*)(ws + off); off += (size_t)2 * D * D * 2;   //   4 MiB
  uint16_t* WoT  = (uint16_t*)(ws + off); off += (size_t)D * D * 2;       //   2 MiB
  uint16_t* attn = (uint16_t*)(ws + off); off += (size_t)B * H * 64 * 64 * 2; // 1 MiB

  dim3 blk(256);

  // 0) one-time bf16 conversion / weight transposes
  cvt_bf16_kernel<<<dim3((M * D / 8 + 255) / 256), blk, 0, stream>>>(x, xb, M * D / 8);
  transpose_cvt_kernel<<<dim3(16, 16), blk, 0, stream>>>(Wqk, BtQV,         D);
  transpose_cvt_kernel<<<dim3(16, 16), blk, 0, stream>>>(Wv,  BtQV + D * D, D);
  transpose_cvt_kernel<<<dim3(16, 16), blk, 0, stream>>>(Wo,  WoT,          D);

  // 1) fused projection: QKV[M x 2048] = xb @ [Wqk | Wv]
  gemm_wmma_kernel<false><<<dim3(2 * D / 128, M / 128), blk, 0, stream>>>(
      xb, BtQV, QKV, M, 2 * D, D);

  // 2) gram + softmax per (b,h)
  gram_softmax_kernel<<<dim3(B * H), blk, 0, stream>>>(QKV, tau, attn);

  // 3) Y = attn applied to V
  apply_attn_kernel<<<dim3(N / 128, B * H), blk, 0, stream>>>(QKV, attn, Ybuf);

  // 4) out = Y @ Wo  (f32 store to d_out)
  gemm_wmma_kernel<true><<<dim3(D / 128, M / 128), blk, 0, stream>>>(
      Ybuf, WoT, d_out, M, D, D);
}
